// MultiHeadAttention_90065464197793
// MI455X (gfx1250) — compile-verified, matrix operand order checked by reading, not run on hardware
//
#include <hip/hip_runtime.h>

typedef __bf16 bf16;
typedef __attribute__((ext_vector_type(16))) __bf16 v16bf;
typedef __attribute__((ext_vector_type(8)))  __bf16 v8bf;
typedef __attribute__((ext_vector_type(8)))  float  v8f;
typedef __attribute__((ext_vector_type(4)))  unsigned int v4u;
typedef __attribute__((ext_vector_type(8)))  unsigned int v8u;

#define HIDDEN 1024
#define NH 16
#define HD 64
#define NB 2
#define NS 2048

// ---------------------------------------------------------------------------
// WMMA helper (gfx1250: D = A(16x32 bf16) * B(32x16 bf16) + C(16x16 f32))
// ---------------------------------------------------------------------------
__device__ __forceinline__ v8f wmma_bf16(v16bf a, v16bf b, v8f c) {
  return __builtin_amdgcn_wmma_f32_16x16x32_bf16(
      /*neg_a=*/false, a, /*neg_b=*/false, b,
      /*c_mod=*/(short)0, c, /*reuse_a=*/false, /*reuse_b=*/false);
}

// Load a 16x32 bf16 fragment from a row-major matrix (row stride `ld` elems).
// Works for A (row = M row) and B (row = N column of Y; all GEMMs are X*Y^T).
// Per-lane layout (ISA 7.12.2, 16-bit A 16x32):
//   lanes 0-15 : row = lane,     K chunks [0..7]  and [16..23]
//   lanes 16-31: row = lane-16,  K chunks [8..15] and [24..31]
__device__ __forceinline__ v16bf load_frag(const bf16* base, int ld) {
  const int lane = threadIdx.x & 31;
  const int row  = lane & 15;
  const int c0   = (lane >> 4) ? 8 : 0;
  const bf16* p  = base + (size_t)row * ld + c0;
  union { v16bf v; v8bf h[2]; } u;
  u.h[0] = *(const v8bf*)(p);
  u.h[1] = *(const v8bf*)(p + 16);
  return u.v;
}

// ---------------------------------------------------------------------------
// f32 -> bf16 conversion
// ---------------------------------------------------------------------------
__global__ void cvt_f32_bf16(const float* __restrict__ src, bf16* __restrict__ dst, int n) {
  int i = blockIdx.x * blockDim.x + threadIdx.x;
  if (i < n) dst[i] = (bf16)src[i];
}

// ---------------------------------------------------------------------------
// Generic GEMM: C[M,N] = X[M,K] * Y[N,K]^T  (bf16 in, bf16 or f32 out)
// One wave computes a 16x64 strip (4 N-tiles), 8 waves / block.
// Register double-buffered K-loop so loads overlap the WMMA pipe.
// ---------------------------------------------------------------------------
__global__ __launch_bounds__(256)
void gemm_xyt(const bf16* __restrict__ X, const bf16* __restrict__ Y,
              bf16* __restrict__ outb, float* __restrict__ outf,
              int M, int N, int K) {
  const int wid     = blockIdx.x * (blockDim.x >> 5) + (threadIdx.x >> 5);
  const int ntiles  = N >> 6;
  const int nstrips = (M >> 4) * ntiles;
  if (wid >= nstrips) return;           // wave-uniform
  const int mt = wid / ntiles;
  const int n0 = (wid % ntiles) << 6;

  const bf16* Xrow = X + (size_t)(mt * 16) * K;
  const bf16* Y0   = Y + (size_t)(n0 +  0) * K;
  const bf16* Y1   = Y + (size_t)(n0 + 16) * K;
  const bf16* Y2   = Y + (size_t)(n0 + 32) * K;
  const bf16* Y3   = Y + (size_t)(n0 + 48) * K;

  v16bf a  = load_frag(Xrow, K);
  v16bf b0 = load_frag(Y0, K);
  v16bf b1 = load_frag(Y1, K);
  v16bf b2 = load_frag(Y2, K);
  v16bf b3 = load_frag(Y3, K);
  v8f acc[4] = {};

  for (int kk = 0; kk < K; kk += 32) {
    const int kn = (kk + 32 < K) ? kk + 32 : 0;   // wrap on last iter (unused)
    if (kk + 160 < K) __builtin_prefetch(Xrow + kk + 160, 0, 1);
    v16bf an  = load_frag(Xrow + kn, K);
    v16bf bn0 = load_frag(Y0 + kn, K);
    v16bf bn1 = load_frag(Y1 + kn, K);
    v16bf bn2 = load_frag(Y2 + kn, K);
    v16bf bn3 = load_frag(Y3 + kn, K);
    acc[0] = wmma_bf16(a, b0, acc[0]);
    acc[1] = wmma_bf16(a, b1, acc[1]);
    acc[2] = wmma_bf16(a, b2, acc[2]);
    acc[3] = wmma_bf16(a, b3, acc[3]);
    a = an; b0 = bn0; b1 = bn1; b2 = bn2; b3 = bn3;
  }

  // C layout: VGPR r, lanes 0-15 -> (row r, col lane); lanes 16-31 -> (row 8+r)
  const int lane = threadIdx.x & 31;
  const int col  = lane & 15;
  const int r0   = (lane >> 4) << 3;
#pragma unroll
  for (int t = 0; t < 4; ++t)
#pragma unroll
    for (int r = 0; r < 8; ++r) {
      size_t idx = (size_t)(mt * 16 + r0 + r) * N + n0 + t * 16 + col;
      if (outb) outb[idx] = (bf16)acc[t][r];
      else      outf[idx] = acc[t][r];
    }
}

// ---------------------------------------------------------------------------
// RoPE on q and k, in place. One thread per (b,s,h,d<32) rotation pair.
// ---------------------------------------------------------------------------
__global__ void rope_qk(bf16* __restrict__ q, bf16* __restrict__ k) {
  int idx = blockIdx.x * blockDim.x + threadIdx.x;
  int d = idx & 31;
  int h = (idx >> 5) & (NH - 1);
  int s = (idx >> 9) & (NS - 1);
  int b = idx >> 20;
  if (b >= NB) return;
  size_t base = (size_t)(b * NS + s) * HIDDEN + h * HD + d;
  // inv_freq = 10000^(-d/32) = exp(-d * ln(10000)/32)
  float ang = (float)s * __expf(-(float)d * 0.28782313662425572f);
  float c = __cosf(ang), sn = __sinf(ang);
  float q1 = (float)q[base], q2 = (float)q[base + 32];
  q[base]      = (bf16)(q1 * c - q2 * sn);
  q[base + 32] = (bf16)(q2 * c + q1 * sn);
  float k1 = (float)k[base], k2 = (float)k[base + 32];
  k[base]      = (bf16)(k1 * c - k2 * sn);
  k[base + 32] = (bf16)(k2 * c + k1 * sn);
}

// ---------------------------------------------------------------------------
// Transpose V: [B*S, NH*HD] -> vT[B, NH, HD, S] so P @ V becomes X * Y^T.
// ---------------------------------------------------------------------------
__global__ void transpose_v(const bf16* __restrict__ v, bf16* __restrict__ vT) {
  int idx = blockIdx.x * blockDim.x + threadIdx.x;
  if (idx >= NB * NS * HIDDEN) return;
  int col = idx & (HIDDEN - 1);
  int rw  = idx >> 10;
  int s   = rw & (NS - 1);
  int b   = rw >> 11;
  int h   = col >> 6;
  int d   = col & 63;
  vT[((size_t)(b * NH + h) * HD + d) * NS + s] = v[idx];
}

// ---------------------------------------------------------------------------
// Flash attention. 8 waves/block share one (b,h); consecutive q-blocks.
// Per 32-key step, staged into LDS once per block, double-buffered:
//   K tile (32 keys x 64 d, 4 KB)  -> Tensor Data Mover (TENSORcnt), wave 0
//   V tile (64 d x 32 keys, 4 KB)  -> async-to-LDS loads (ASYNCcnt), all lanes
// LDS map: kt[2] @ 0/4096, vt[2] @ 8192/12288, pstage @ 16384.
// ---------------------------------------------------------------------------
struct AttnSmem {
  bf16 kt[2][32][64];
  bf16 vt[2][64][32];
  bf16 pstage[8][16][32];
};

// TDM descriptor build + issue (ISA ch.8: D# groups in SGPRs, 2D tile).
// dim0 = contiguous elements per row, dim1 = rows, stride0 in elements.
__device__ __forceinline__ void tdm_load_tile(unsigned lds_off, const void* gptr,
                                              unsigned dim0, unsigned dim1,
                                              unsigned stride0) {
  unsigned long long ga = (unsigned long long)gptr;
  v4u g0;
  g0[0] = 1u;                                 // count=1, user descriptor
  g0[1] = lds_off;                            // lds_addr (bytes)
  g0[2] = (unsigned)ga;                       // global_addr[31:0]
  g0[3] = (unsigned)(ga >> 32) | (2u << 30);  // global_addr[56:32] | type=2
  v8u g1;
  g1[0] = 1u << 16;                           // data_size = 2 bytes
  g1[1] = (dim0 & 0xffffu) << 16;             // tensor_dim0[15:0]
  g1[2] = (dim1 & 0xffffu) << 16;             // tensor_dim0 hi | tensor_dim1[15:0]
  g1[3] = (dim0 & 0xffffu) << 16;             // tensor_dim1 hi | tile_dim0
  g1[4] = dim1 & 0xffffu;                     // tile_dim1 | tile_dim2=0
  g1[5] = stride0;                            // tensor_dim0_stride[31:0] (elems)
  g1[6] = 0u;                                 // stride hi | dim1_stride lo
  g1[7] = 0u;
  v4u g2 = {0u, 0u, 0u, 0u};
  v4u g3 = {0u, 0u, 0u, 0u};
  asm volatile("tensor_load_to_lds %0, %1, %2, %3"
               :: "s"(g0), "s"(g1), "s"(g2), "s"(g3) : "memory");
}

// Each of 256 threads copies one 16B chunk of the V tile (64 rows x 4 chunks).
__device__ __forceinline__ void async_copy_v(const bf16* vbase, int key0, int buf, int t) {
  const int vrow = t >> 2, vch = t & 3;
  const bf16* vg = vbase + (size_t)vrow * NS + key0 + vch * 8;
  unsigned vl = (unsigned)(8192 + buf * 4096 + (vrow * 32 + vch * 8) * 2);
  asm volatile("global_load_async_to_lds_b128 %0, %1, off"
               :: "v"(vl), "v"(vg) : "memory");
}

__global__ __launch_bounds__(256)
void attn(const bf16* __restrict__ q, const bf16* __restrict__ k,
          const bf16* __restrict__ vT, bf16* __restrict__ O) {
  __shared__ AttnSmem sm;

  const int tid  = threadIdx.x;
  const int wv   = tid >> 5;
  const int lane = tid & 31;
  const int wid  = blockIdx.x * 8 + wv;

  const int qblk = wid & (NS / 16 - 1);          // low bits -> same (b,h) per block
  const int h    = (wid >> 7) & (NH - 1);
  const int b    = wid >> 11;

  const bf16* qbase = q  + (size_t)(b * NS + qblk * 16) * HIDDEN + h * HD;
  const bf16* kbase = k  + (size_t)b * NS * HIDDEN + h * HD;   // block-uniform
  const bf16* vbase = vT + (size_t)(b * NH + h) * HD * NS;     // block-uniform
  bf16*       obase = O  + (size_t)(b * NS + qblk * 16) * HIDDEN + h * HD;

  const v16bf qa0 = load_frag(qbase, HIDDEN);
  const v16bf qa1 = load_frag(qbase + 32, HIDDEN);

  v8f o0 = {}, o1 = {}, o2 = {}, o3 = {};
  float mrow[8], lrow[8];
#pragma unroll
  for (int r = 0; r < 8; ++r) { mrow[r] = -1e30f; lrow[r] = 0.f; }

  const int   col   = lane & 15;
  const int   r0    = (lane >> 4) << 3;
  const int   pc0   = (lane >> 4) ? 8 : 0;
  const float scale = 0.125f;                    // 1/sqrt(64)
  const int   NSTEP = NS / 32;                   // 64

  // prologue: stage step 0 into buffer 0
  if (wv == 0) tdm_load_tile(0u, kbase, 64u, 32u, (unsigned)HIDDEN);
  async_copy_v(vbase, 0, 0, tid);

  for (int j = 0; j < NSTEP; ++j) {
    const int buf = j & 1;
    if (j > 0) __syncthreads();                  // all waves done with buf^1
    if (j + 1 < NSTEP) {
      async_copy_v(vbase, (j + 1) * 32, buf ^ 1, tid);
      if (wv == 0)
        tdm_load_tile((unsigned)((buf ^ 1) * 4096),
                      kbase + (size_t)(j + 1) * 32 * HIDDEN,
                      64u, 32u, (unsigned)HIDDEN);
      asm volatile("s_wait_asynccnt 0x1" ::: "memory");
      if (wv == 0) asm volatile("s_wait_tensorcnt 0x1" ::: "memory");
    } else {
      asm volatile("s_wait_asynccnt 0x0" ::: "memory");
      if (wv == 0) asm volatile("s_wait_tensorcnt 0x0" ::: "memory");
    }
    __syncthreads();                             // everyone's step-j data landed

    // ---- scores: 16 queries x 32 keys (two 16x16 C tiles), K from LDS ----
    v8f s0 = {}, s1 = {};
    {
      v16bf b00 = load_frag(&sm.kt[buf][0][0],   64);
      v16bf b01 = load_frag(&sm.kt[buf][0][32],  64);
      v16bf b10 = load_frag(&sm.kt[buf][16][0],  64);
      v16bf b11 = load_frag(&sm.kt[buf][16][32], 64);
      s0 = wmma_bf16(qa0, b00, s0);
      s0 = wmma_bf16(qa1, b01, s0);
      s1 = wmma_bf16(qa0, b10, s1);
      s1 = wmma_bf16(qa1, b11, s1);
    }

    // ---- online softmax; row stats replicated across each 16-lane group ----
    float p0[8], p1[8];
#pragma unroll
    for (int r = 0; r < 8; ++r) {
      float a  = s0[r] * scale;
      float bb = s1[r] * scale;
      float mx = fmaxf(a, bb);
#pragma unroll
      for (int m = 1; m <= 8; m <<= 1)
        mx = fmaxf(mx, __shfl_xor(mx, m, 32));
      float mnew = fmaxf(mrow[r], mx);
      float corr = __expf(mrow[r] - mnew);
      float e0 = __expf(a - mnew);
      float e1 = __expf(bb - mnew);
      float rs = e0 + e1;
#pragma unroll
      for (int m = 1; m <= 8; m <<= 1)
        rs += __shfl_xor(rs, m, 32);
      lrow[r] = lrow[r] * corr + rs;
      mrow[r] = mnew;
      p0[r] = e0;
      p1[r] = e1;
      o0[r] *= corr; o1[r] *= corr; o2[r] *= corr; o3[r] *= corr;
    }

    // ---- stage P (16x32 bf16) through LDS: C-layout -> A-layout ----
#pragma unroll
    for (int r = 0; r < 8; ++r) {
      sm.pstage[wv][r0 + r][col]      = (bf16)p0[r];
      sm.pstage[wv][r0 + r][16 + col] = (bf16)p1[r];
    }
    asm volatile("s_wait_dscnt 0x0" ::: "memory");  // LDS RAW, same wave

    v16bf pa;
    {
      const bf16* pp = &sm.pstage[wv][lane & 15][pc0];
      union { v16bf v; v8bf hh[2]; } u;
      u.hh[0] = *(const v8bf*)(pp);
      u.hh[1] = *(const v8bf*)(pp + 16);
      pa = u.v;
    }

    // ---- O(16x64) += P(16x32) @ V(32x64), V fragments from LDS ----
    o0 = wmma_bf16(pa, load_frag(&sm.vt[buf][0][0],  32), o0);
    o1 = wmma_bf16(pa, load_frag(&sm.vt[buf][16][0], 32), o1);
    o2 = wmma_bf16(pa, load_frag(&sm.vt[buf][32][0], 32), o2);
    o3 = wmma_bf16(pa, load_frag(&sm.vt[buf][48][0], 32), o3);
  }

  // ---- normalize and store O tile (bf16) ----
#pragma unroll
  for (int r = 0; r < 8; ++r) {
    float inv = 1.0f / lrow[r];
    size_t row = (size_t)(r0 + r) * HIDDEN;
    obase[row + col]      = (bf16)(o0[r] * inv);
    obase[row + 16 + col] = (bf16)(o1[r] * inv);
    obase[row + 32 + col] = (bf16)(o2[r] * inv);
    obase[row + 48 + col] = (bf16)(o3[r] * inv);
  }
}

// ---------------------------------------------------------------------------
// Host-side launcher
// ---------------------------------------------------------------------------
extern "C" void kernel_launch(void* const* d_in, const int* in_sizes, int n_in,
                              void* d_out, int out_size, void* d_ws, size_t ws_size,
                              hipStream_t stream) {
  const float* x  = (const float*)d_in[0];
  const float* Wq = (const float*)d_in[1];
  const float* Wk = (const float*)d_in[2];
  const float* Wv = (const float*)d_in[3];
  const float* Wo = (const float*)d_in[4];
  float* out = (float*)d_out;

  char* ws = (char*)d_ws;
  const size_t XB = (size_t)NB * NS * HIDDEN * 2;   // 8 MiB
  const size_t WB = (size_t)HIDDEN * HIDDEN * 2;    // 2 MiB
  bf16* xb  = (bf16*)(ws);
  bf16* wqb = (bf16*)(ws + XB);
  bf16* wkb = (bf16*)(ws + XB + WB);
  bf16* wvb = (bf16*)(ws + XB + 2 * WB);
  bf16* wob = (bf16*)(ws + XB + 3 * WB);
  bf16* qb  = (bf16*)(ws + XB + 4 * WB);
  bf16* kb  = (bf16*)(ws + 2 * XB + 4 * WB);
  bf16* vb  = (bf16*)(ws + 3 * XB + 4 * WB);
  bf16* vtb = (bf16*)(ws + 4 * XB + 4 * WB);
  bf16* ob  = (bf16*)(ws + 5 * XB + 4 * WB);

  const int M  = NB * NS;        // 4096
  const int NX = M * HIDDEN;     // 4194304
  const int NW = HIDDEN * HIDDEN;

  cvt_f32_bf16<<<NX / 256, 256, 0, stream>>>(x,  xb,  NX);
  cvt_f32_bf16<<<NW / 256, 256, 0, stream>>>(Wq, wqb, NW);
  cvt_f32_bf16<<<NW / 256, 256, 0, stream>>>(Wk, wkb, NW);
  cvt_f32_bf16<<<NW / 256, 256, 0, stream>>>(Wv, wvb, NW);
  cvt_f32_bf16<<<NW / 256, 256, 0, stream>>>(Wo, wob, NW);

  const int strips = (M / 16) * (HIDDEN / 64);      // 4096
  const int gblk   = strips / 8;                    // 512 blocks of 8 waves
  gemm_xyt<<<gblk, 256, 0, stream>>>(xb, wqb, qb, nullptr, M, HIDDEN, HIDDEN);
  gemm_xyt<<<gblk, 256, 0, stream>>>(xb, wkb, kb, nullptr, M, HIDDEN, HIDDEN);
  gemm_xyt<<<gblk, 256, 0, stream>>>(xb, wvb, vb, nullptr, M, HIDDEN, HIDDEN);

  rope_qk<<<(NB * NS * NH * 32) / 256, 256, 0, stream>>>(qb, kb);
  transpose_v<<<NX / 256, 256, 0, stream>>>(vb, vtb);

  const int awaves = NB * NH * (NS / 16);           // 4096 waves
  attn<<<awaves / 8, 256, 0, stream>>>(qb, kb, vtb, ob);

  gemm_xyt<<<gblk, 256, 0, stream>>>(ob, wob, nullptr, out, M, HIDDEN, HIDDEN);
}
